// QuantLlamaAttention_28905129902562
// MI455X (gfx1250) — matrix operand order repaired and checked by reading, not compiled, not run on hardware
//
#include <hip/hip_runtime.h>
#include <cstdint>

#define B_  2
#define S_  1024
#define D_  4096
#define H_  32
#define HD_ 128
#define M_  (B_*S_)   // 2048 token rows

typedef __attribute__((ext_vector_type(16))) __bf16 v16bf;
typedef __attribute__((ext_vector_type(8)))  float  v8f;
typedef __attribute__((ext_vector_type(8)))  short  v8s;
typedef __attribute__((ext_vector_type(16))) short  v16s;
typedef __attribute__((ext_vector_type(4)))  int    v4i;

union AFrag { v16s s; v16bf b; struct { v8s lo, hi; } p; };
union BFrag { v16s s; v16bf b; };

// ---------- gfx1250 async global->LDS copy (builtin confirmed present) ----------
#if defined(__has_builtin)
#if __has_builtin(__builtin_amdgcn_global_load_async_to_lds_b128)
#define HAVE_ASYNC_LDS 1
#endif
#endif
#ifndef HAVE_ASYNC_LDS
#define HAVE_ASYNC_LDS 0
#warning "gfx1250 async-to-LDS builtin not found; using synchronous LDS staging fallback"
#endif

typedef __attribute__((address_space(1))) v4i as1_v4i;   // global (prints as __device__)
typedef __attribute__((address_space(3))) v4i as3_v4i;   // LDS

__device__ __forceinline__ void async_cp_b128(const unsigned short* g, unsigned short* l) {
#if HAVE_ASYNC_LDS
    __builtin_amdgcn_global_load_async_to_lds_b128(
        (as1_v4i*)(uintptr_t)g,
        (as3_v4i*)(unsigned int)(uintptr_t)l,
        0, 0);
#else
    *(v8s*)l = *(const v8s*)g;
#endif
}

__device__ __forceinline__ void async_wait0() {
#if HAVE_ASYNC_LDS
#if __has_builtin(__builtin_amdgcn_s_wait_asynccnt)
    __builtin_amdgcn_s_wait_asynccnt(0);
#else
    asm volatile("s_wait_asynccnt 0x0" ::: "memory");
#endif
#endif
}

__device__ __forceinline__ unsigned short f2bf(float f) {
    unsigned u = __float_as_uint(f);
    u += 0x7fffu + ((u >> 16) & 1u);          // RTNE
    return (unsigned short)(u >> 16);
}

__device__ __forceinline__ float fq(float x, float scale) {
    float q = rintf(x / scale);
    q = fminf(fmaxf(q, -127.f), 127.f);
    return q * scale;
}

// ---------------- fp32 -> bf16 convert ----------------
__global__ __launch_bounds__(256) void to_bf16(const float* __restrict__ in,
                                               unsigned short* __restrict__ out, int n) {
    int i = blockIdx.x * 256 + threadIdx.x;
    if (i < n) out[i] = f2bf(in[i]);
}

// ---------------- quantize fp32 -> bf16 with global scale ----------------
__global__ __launch_bounds__(256) void quant_to_bf16(const float* __restrict__ in,
                                                     unsigned short* __restrict__ out,
                                                     const unsigned* __restrict__ amax, int n) {
    int i = blockIdx.x * 256 + threadIdx.x;
    if (i >= n) return;
    float s = fmaxf(__uint_as_float(*amax), 1e-8f) / 127.f;
    out[i] = f2bf(fq(in[i], s));
}

// ---------------- C[M,N] = A[M,K](bf16) * W[N,K]^T(bf16), fp32 accum ----------------
// 8 waves, block tile 128x128, K-step 32, double-buffered LDS staged via async copies.
// Wave (wm: 0..3, wn: 0..1) computes a 32x64 strip = 2x4 WMMA tiles.
#define BK      32
#define STRIDE  40                 // padded LDS row stride (elements) to spread banks
#define TILE_E  (128 * STRIDE)     // elements per LDS tile buffer

__global__ __launch_bounds__(256) void gemm_bf16_nt(const unsigned short* __restrict__ A,
                                                    const unsigned short* __restrict__ W,
                                                    float* __restrict__ C,
                                                    int Mdim, int Ndim, int Kdim) {
    __shared__ __attribute__((aligned(32))) unsigned short As[2 * TILE_E];
    __shared__ __attribute__((aligned(32))) unsigned short Bs[2 * TILE_E];

    const int tid   = threadIdx.x;
    const int lane  = tid & 31;
    const int wave  = tid >> 5;
    const int lmod  = lane & 15;
    const int lhalf = lane >> 4;
    const int wm    = wave >> 1;           // 0..3
    const int wn    = wave & 1;            // 0..1
    const int tm0   = blockIdx.y * 128;
    const int tn0   = blockIdx.x * 128;
    (void)Mdim;

    // copy assignment: thread -> (row 0..127, 32B segment 0..1) of the 128x32 tile
    const int crow = tid >> 1;
    const int cseg = tid & 1;
    const unsigned short* ag = A + (size_t)(tm0 + crow) * Kdim + cseg * 16;
    const unsigned short* wg = W + (size_t)(tn0 + crow) * Kdim + cseg * 16;
    unsigned short* al = As + crow * STRIDE + cseg * 16;
    unsigned short* bl = Bs + crow * STRIDE + cseg * 16;

    v8f acc[2][4] = {};

    const int NK = Kdim / BK;
    // prologue: stage tile 0 into buffer 0
    {
        async_cp_b128(ag, al);      async_cp_b128(ag + 8, al + 8);
        async_cp_b128(wg, bl);      async_cp_b128(wg + 8, bl + 8);
    }

    for (int it = 0; it < NK; ++it) {
        const int cur = it & 1;
        async_wait0();
        __syncthreads();                       // tile 'cur' resident for all waves
        if (it + 1 < NK) {                     // stage next tile into other buffer
            const int nk = (it + 1) * BK;
            async_cp_b128(ag + nk,     al + (cur ^ 1) * TILE_E);
            async_cp_b128(ag + nk + 8, al + (cur ^ 1) * TILE_E + 8);
            async_cp_b128(wg + nk,     bl + (cur ^ 1) * TILE_E);
            async_cp_b128(wg + nk + 8, bl + (cur ^ 1) * TILE_E + 8);
        }

        const unsigned short* as = As + cur * TILE_E;
        const unsigned short* bs = Bs + cur * TILE_E;
        AFrag a0, a1;
        const int ar = wm * 32 + lmod;
        a0.p.lo = *(const v8s*)(as + ar * STRIDE + lhalf * 16);
        a0.p.hi = *(const v8s*)(as + ar * STRIDE + lhalf * 16 + 8);
        a1.p.lo = *(const v8s*)(as + (ar + 16) * STRIDE + lhalf * 16);
        a1.p.hi = *(const v8s*)(as + (ar + 16) * STRIDE + lhalf * 16 + 8);
#pragma unroll
        for (int n = 0; n < 4; n++) {
            BFrag bb;
            const int br = wn * 64 + n * 16 + lmod;
            bb.s = *(const v16s*)(bs + br * STRIDE + lhalf * 16);
            acc[0][n] = __builtin_amdgcn_wmma_f32_16x16x32_bf16(false, a0.b, false, bb.b, (short)0, acc[0][n], false, false);
            acc[1][n] = __builtin_amdgcn_wmma_f32_16x16x32_bf16(false, a1.b, false, bb.b, (short)0, acc[1][n], false, false);
        }
        __syncthreads();                       // all waves done with 'cur' before overwrite
    }

#pragma unroll
    for (int m = 0; m < 2; m++) {
        float* crowp = C + (size_t)(tm0 + wm * 32 + m * 16 + lhalf * 8) * Ndim + tn0 + wn * 64 + lmod;
#pragma unroll
        for (int n = 0; n < 4; n++) {
#pragma unroll
            for (int r = 0; r < 8; r++) {
                crowp[(size_t)r * Ndim + n * 16] = acc[m][n][r];
            }
        }
    }
}

// ---------------- RoPE (in place) + global absmax of Q,K,V ----------------
__global__ __launch_bounds__(256) void rope_stats(float* __restrict__ Q, float* __restrict__ K,
                                                  const float* __restrict__ V,
                                                  const int* __restrict__ pos,
                                                  unsigned* __restrict__ amax) {
    int i  = blockIdx.x * 256 + threadIdx.x;     // B*S*H*64 threads
    int hd = i & 63;
    int h  = (i >> 6)  & (H_ - 1);
    int s  = (i >> 11) & (S_ - 1);
    int b  = i >> 21;
    size_t base = ((size_t)(b * S_ + s)) * D_ + h * HD_;

    float p = (float)pos[b * S_ + s];
    float inv_freq = __powf(10000.0f, -(float)(2 * hd) / (float)HD_);
    float f = p * inv_freq, sn, cs;
    __sincosf(f, &sn, &cs);

    float q1 = Q[base + hd], q2 = Q[base + 64 + hd];
    float k1 = K[base + hd], k2 = K[base + 64 + hd];
    float qr1 = q1 * cs - q2 * sn, qr2 = q2 * cs + q1 * sn;
    float kr1 = k1 * cs - k2 * sn, kr2 = k2 * cs + k1 * sn;
    Q[base + hd] = qr1; Q[base + 64 + hd] = qr2;
    K[base + hd] = kr1; K[base + 64 + hd] = kr2;

    float mq = fmaxf(fabsf(qr1), fabsf(qr2));
    float mk = fmaxf(fabsf(kr1), fabsf(kr2));
    float mv = fmaxf(fabsf(V[base + hd]), fabsf(V[base + 64 + hd]));
    for (int m = 16; m > 0; m >>= 1) {
        mq = fmaxf(mq, __shfl_xor(mq, m, 32));
        mk = fmaxf(mk, __shfl_xor(mk, m, 32));
        mv = fmaxf(mv, __shfl_xor(mv, m, 32));
    }
    if ((threadIdx.x & 31) == 0) {
        atomicMax(&amax[0], __float_as_uint(mq));
        atomicMax(&amax[1], __float_as_uint(mk));
        atomicMax(&amax[2], __float_as_uint(mv));
    }
}

// ---------------- fake-quant Q,K,V -> bf16, head-major; V transposed ----------------
__global__ __launch_bounds__(256) void quant_pack(const float* __restrict__ Q,
                                                  const float* __restrict__ K,
                                                  const float* __restrict__ V,
                                                  const unsigned* __restrict__ amax,
                                                  unsigned short* __restrict__ Qb,
                                                  unsigned short* __restrict__ Kb,
                                                  unsigned short* __restrict__ Vtb) {
    int i  = blockIdx.x * 256 + threadIdx.x;     // B*S*H*HD threads
    int hd = i & (HD_ - 1);
    int h  = (i >> 7)  & (H_ - 1);
    int s  = (i >> 12) & (S_ - 1);
    int b  = i >> 22;
    size_t src = ((size_t)(b * S_ + s)) * D_ + h * HD_ + hd;

    float sq = fmaxf(__uint_as_float(amax[0]), 1e-8f) / 127.f;
    float sk = fmaxf(__uint_as_float(amax[1]), 1e-8f) / 127.f;
    float sv = fmaxf(__uint_as_float(amax[2]), 1e-8f) / 127.f;

    size_t dst = (((size_t)(b * H_ + h)) * S_ + s) * HD_ + hd;   // (B,H,S,HD)
    Qb[dst] = f2bf(fq(Q[src], sq));
    Kb[dst] = f2bf(fq(K[src], sk));
    size_t dvt = (((size_t)(b * H_ + h)) * HD_ + hd) * S_ + s;   // (B,H,HD,S)
    Vtb[dvt] = f2bf(fq(V[src], sv));
}

// ---------------- global absmax of QK^T/sqrt(HD) (pre-mask, per reference) ----------------
__global__ __launch_bounds__(32) void score_absmax(const unsigned short* __restrict__ Qb,
                                                   const unsigned short* __restrict__ Kb,
                                                   unsigned* __restrict__ amax) {
    const int lane = threadIdx.x & 31;
    const int lmod = lane & 15, lhalf = lane >> 4;
    int qt = blockIdx.x & 63;
    int h  = (blockIdx.x >> 6) & (H_ - 1);
    int b  = blockIdx.x >> 11;
    const size_t bh = (size_t)(b * H_ + h);

    const unsigned short* qrow = Qb + (bh * S_ + qt * 16 + lmod) * HD_;
    v16bf qa[4];
#pragma unroll
    for (int f = 0; f < 4; f++) {
        AFrag u;
        u.p.lo = *(const v8s*)(qrow + f * 32 + lhalf * 16);
        u.p.hi = *(const v8s*)(qrow + f * 32 + lhalf * 16 + 8);
        qa[f] = u.b;
    }

    const float inv_sqrt_hd = 0.08838834764831845f;   // 1/sqrt(128)
    float lmax = 0.f;
    for (int jt = 0; jt < S_ / 16; jt++) {
        const unsigned short* krow = Kb + (bh * S_ + jt * 16 + lmod) * HD_ + lhalf * 16;
        v8f acc = {};
#pragma unroll
        for (int f = 0; f < 4; f++) {
            BFrag u; u.s = *(const v16s*)(krow + f * 32);
            acc = __builtin_amdgcn_wmma_f32_16x16x32_bf16(false, qa[f], false, u.b, (short)0, acc, false, false);
        }
#pragma unroll
        for (int r = 0; r < 8; r++) lmax = fmaxf(lmax, fabsf(acc[r] * inv_sqrt_hd));
    }
    for (int m = 16; m > 0; m >>= 1) lmax = fmaxf(lmax, __shfl_xor(lmax, m, 32));
    if (lane == 0) atomicMax(&amax[3], __float_as_uint(lmax));
}

// ---------------- attention: one wave per (b, h, 16 query rows) ----------------
// LDS: 16x1024 fp32 scores (64KB) + 16x1024 bf16 probs (32KB) = 96KB/workgroup.
__global__ __launch_bounds__(32) void attention_kernel(const unsigned short* __restrict__ Qb,
                                                       const unsigned short* __restrict__ Kb,
                                                       const unsigned short* __restrict__ Vtb,
                                                       const unsigned* __restrict__ amax,
                                                       float* __restrict__ AO,
                                                       unsigned* __restrict__ amax_o) {
    extern __shared__ char smem[];
    float*          sc = (float*)smem;                                      // [16][S_]
    unsigned short* pb = (unsigned short*)(smem + 16 * S_ * sizeof(float)); // [16][S_]

    const int lane = threadIdx.x & 31;
    const int lmod = lane & 15, lhalf = lane >> 4;
    int qt = blockIdx.x & 63;
    int h  = (blockIdx.x >> 6) & (H_ - 1);
    int b  = blockIdx.x >> 11;
    const size_t bh = (size_t)(b * H_ + h);

    const unsigned short* qrow = Qb + (bh * S_ + qt * 16 + lmod) * HD_;
    v16bf qa[4];
#pragma unroll
    for (int f = 0; f < 4; f++) {
        AFrag u;
        u.p.lo = *(const v8s*)(qrow + f * 32 + lhalf * 16);
        u.p.hi = *(const v8s*)(qrow + f * 32 + lhalf * 16 + 8);
        qa[f] = u.b;
    }

    const float inv_sqrt_hd = 0.08838834764831845f;
    const float sscale = fmaxf(__uint_as_float(amax[3]), 1e-8f) / 127.f;

    // scores -> quantize -> mask -> LDS
    for (int jt = 0; jt < S_ / 16; jt++) {
        const unsigned short* krow = Kb + (bh * S_ + jt * 16 + lmod) * HD_ + lhalf * 16;
        v8f acc = {};
#pragma unroll
        for (int f = 0; f < 4; f++) {
            BFrag u; u.s = *(const v16s*)(krow + f * 32);
            acc = __builtin_amdgcn_wmma_f32_16x16x32_bf16(false, qa[f], false, u.b, (short)0, acc, false, false);
        }
#pragma unroll
        for (int r = 0; r < 8; r++) {
            int row  = r + 8 * lhalf;
            int col  = jt * 16 + lmod;
            int grow = qt * 16 + row;
            float sv = fq(acc[r] * inv_sqrt_hd, sscale);
            if (col > grow) sv = -1.0e9f;
            sc[row * S_ + col] = sv;
        }
    }

    // exact softmax per row + prob quantization (global probs scale is exactly 1/127)
    const float pscale = 1.0f / 127.0f;
    for (int row = 0; row < 16; row++) {
        float m = -3.0e38f;
        for (int c = lane; c < S_; c += 32) m = fmaxf(m, sc[row * S_ + c]);
        for (int t = 16; t > 0; t >>= 1) m = fmaxf(m, __shfl_xor(m, t, 32));
        float sum = 0.f;
        for (int c = lane; c < S_; c += 32) {
            float e = __expf(sc[row * S_ + c] - m);
            sc[row * S_ + c] = e;
            sum += e;
        }
        for (int t = 16; t > 0; t >>= 1) sum += __shfl_xor(sum, t, 32);
        float inv = 1.0f / sum;
        for (int c = lane; c < S_; c += 32) {
            float pq = rintf(sc[row * S_ + c] * inv * 127.0f) * pscale;
            pb[row * S_ + c] = f2bf(pq);
        }
    }

    // out(16x128) = P(16x1024, LDS) x V^T(B,H,HD,S)
    v8f oacc[8] = {};
    for (int kt = 0; kt < S_; kt += 32) {
        AFrag a;
        a.p.lo = *(const v8s*)(pb + lmod * S_ + kt + lhalf * 16);
        a.p.hi = *(const v8s*)(pb + lmod * S_ + kt + lhalf * 16 + 8);
#pragma unroll
        for (int sub = 0; sub < 8; sub++) {
            const unsigned short* vrow = Vtb + (bh * HD_ + sub * 16 + lmod) * S_ + kt + lhalf * 16;
            BFrag u; u.s = *(const v16s*)(vrow);
            oacc[sub] = __builtin_amdgcn_wmma_f32_16x16x32_bf16(false, a.b, false, u.b, (short)0, oacc[sub], false, false);
        }
    }

    float lmax = 0.f;
    int grow = qt * 16 + 8 * lhalf;
#pragma unroll
    for (int sub = 0; sub < 8; sub++) {
#pragma unroll
        for (int r = 0; r < 8; r++) {
            float v = oacc[sub][r];
            size_t idx = ((size_t)(b * S_ + grow + r)) * D_ + h * HD_ + sub * 16 + lmod;
            AO[idx] = v;
            lmax = fmaxf(lmax, fabsf(v));
        }
    }
    for (int t = 16; t > 0; t >>= 1) lmax = fmaxf(lmax, __shfl_xor(lmax, t, 32));
    if (lane == 0) atomicMax(amax_o, __float_as_uint(lmax));
}

// ======================================================================
extern "C" void kernel_launch(void* const* d_in, const int* in_sizes, int n_in,
                              void* d_out, int out_size, void* d_ws, size_t ws_size,
                              hipStream_t stream) {
    (void)in_sizes; (void)n_in; (void)out_size; (void)ws_size;
    const float* hs = (const float*)d_in[0];
    const float* wq = (const float*)d_in[1];
    const float* wk = (const float*)d_in[2];
    const float* wv = (const float*)d_in[3];
    const float* wo = (const float*)d_in[4];
    const int*  pos = (const int*)d_in[5];
    float* out = (float*)d_out;

    char* ws = (char*)d_ws;
    unsigned* amax = (unsigned*)ws;                 // [0]=q [1]=k [2]=v [3]=scores [4]=attn_out
    size_t off = 256;
    const size_t SZ_XB = (size_t)M_ * D_ * 2;
    const size_t SZ_WB = (size_t)D_ * D_ * 2;
    const size_t SZ_F  = (size_t)M_ * D_ * 4;
    const size_t SZ_HB = (size_t)M_ * D_ * 2;

    unsigned short* Xb  = (unsigned short*)(ws + off); off += SZ_XB;
    unsigned short* Wqb = (unsigned short*)(ws + off); off += SZ_WB;
    unsigned short* Wkb = (unsigned short*)(ws + off); off += SZ_WB;
    unsigned short* Wvb = (unsigned short*)(ws + off); off += SZ_WB;
    unsigned short* Wob = (unsigned short*)(ws + off); off += SZ_WB;
    float* Qf  = (float*)(ws + off); off += SZ_F;
    float* Kf  = (float*)(ws + off); off += SZ_F;
    float* Vf  = (float*)(ws + off); off += SZ_F;
    unsigned short* Qb  = (unsigned short*)(ws + off); off += SZ_HB;
    unsigned short* Kb  = (unsigned short*)(ws + off); off += SZ_HB;
    unsigned short* Vtb = (unsigned short*)(ws + off); off += SZ_HB;
    float* AOf = (float*)(ws + off); off += SZ_F;
    unsigned short* AOb = (unsigned short*)(ws + off); off += SZ_HB;

    (void)hipMemsetAsync(ws, 0, 256, stream);       // zero absmax accumulators

    const int nX = M_ * D_, nW = D_ * D_;
    to_bf16<<<nX / 256, 256, 0, stream>>>(hs, Xb, nX);
    to_bf16<<<nW / 256, 256, 0, stream>>>(wq, Wqb, nW);
    to_bf16<<<nW / 256, 256, 0, stream>>>(wk, Wkb, nW);
    to_bf16<<<nW / 256, 256, 0, stream>>>(wv, Wvb, nW);
    to_bf16<<<nW / 256, 256, 0, stream>>>(wo, Wob, nW);

    dim3 gg(D_ / 128, M_ / 128);                    // 32 x 16 blocks, 256 thr
    gemm_bf16_nt<<<gg, 256, 0, stream>>>(Xb, Wqb, Qf, M_, D_, D_);
    gemm_bf16_nt<<<gg, 256, 0, stream>>>(Xb, Wkb, Kf, M_, D_, D_);
    gemm_bf16_nt<<<gg, 256, 0, stream>>>(Xb, Wvb, Vf, M_, D_, D_);

    rope_stats<<<(B_ * S_ * H_ * 64) / 256, 256, 0, stream>>>(Qf, Kf, Vf, pos, amax);
    quant_pack<<<(B_ * S_ * D_) / 256, 256, 0, stream>>>(Qf, Kf, Vf, amax, Qb, Kb, Vtb);

    const int nBHQ = B_ * H_ * (S_ / 16);           // 4096 waves
    score_absmax<<<nBHQ, 32, 0, stream>>>(Qb, Kb, amax);
    const size_t lds_bytes = 16 * S_ * sizeof(float) + 16 * S_ * sizeof(unsigned short); // 96KB
    attention_kernel<<<nBHQ, 32, lds_bytes, stream>>>(Qb, Kb, Vtb, amax, AOf, amax + 4);

    quant_to_bf16<<<nX / 256, 256, 0, stream>>>(AOf, AOb, amax + 4, nX);
    gemm_bf16_nt<<<gg, 256, 0, stream>>>(AOb, Wob, out, M_, D_, D_);
}
